// INT8Linear_44899588112657
// MI455X (gfx1250) — compile-verified, matrix operand order checked by reading, not run on hardware
//
#include <hip/hip_runtime.h>
#include <hip/hip_bf16.h>

// ---------------------------------------------------------------------------
// INT8-weight linear, MI455X / gfx1250, wave32.
//   out[32,8192] = x[32,8192] @ (w_int8[8192,8192]*scale).T + bias
//
// Memory-bound on the 64MB int8 weight stream (~2.9us at 23.3 TB/s).
// Fast path (3 kernels, needs 9MB workspace):
//   1) prep:   x bf16 -> f16 (exact), plus per-token row sums of x
//   2) main:   split-K=8, N=32/wave, 2048 waves; int8 -> f16 dequant via
//              v_xor_b32 + v_perm_b32 (value = 1024 + (v+128), exact in f16);
//              v_wmma_f32_16x16x32_f16; f32 partials to workspace
//   3) reduce: fixed-order partial sum, subtract 1152 * rowsum(x) (folds the
//              dequant bias out of the inner loop), * scale, + bias, -> bf16
// Fallback (small ws): single-kernel bf16 dequant WMMA path.
// ---------------------------------------------------------------------------

typedef __attribute__((ext_vector_type(16))) _Float16 v16h;
typedef __attribute__((ext_vector_type(16))) __bf16   v16bf;
typedef __attribute__((ext_vector_type(8)))  float    v8f;

#define K_DIM 8192
#define N_DIM 8192
#define M_DIM 32
#define SPLITK 8
#define K_CHUNK (K_DIM / SPLITK)          // 1024
#define N_PER_WAVE 32
#define WAVES_PER_BLOCK 4
#define THREADS_PER_BLOCK (WAVES_PER_BLOCK * 32)

// workspace layout (bytes)
#define WS_XF16 0                         // 32*8192 f16   = 512KB
#define WS_XSUM (512 * 1024)              // 32 f32
#define WS_PART (1024 * 1024)             // SPLITK*32*8192 f32 = 8MB
#define WS_NEEDED (WS_PART + (size_t)SPLITK * M_DIM * N_DIM * 4)

// ---------------------------------------------------------------------------
// Kernel 1: x bf16 -> f16 (exact) + per-row sums (for dequant-bias folding)
// ---------------------------------------------------------------------------
__global__ __launch_bounds__(256)
void prep_kernel(const __hip_bfloat16* __restrict__ x,
                 _Float16* __restrict__ xf16,
                 float* __restrict__ xsum)
{
    __shared__ float red[256];
    const int m = blockIdx.x;          // 0..31 (token)
    const int t = threadIdx.x;         // 0..255

    const __hip_bfloat16* row  = x    + (size_t)m * K_DIM;
    _Float16*             orow = xf16 + (size_t)m * K_DIM;

    float s = 0.0f;
    for (int k = t; k < K_DIM; k += 256) {
        _Float16 h = (_Float16)(float)row[k];
        orow[k] = h;
        s += (float)h;
    }
    red[t] = s;
    __syncthreads();
    for (int off = 128; off > 0; off >>= 1) {   // fixed-order tree: deterministic
        if (t < off) red[t] += red[t + off];
        __syncthreads();
    }
    if (t == 0) xsum[m] = red[0];
}

// ---------------------------------------------------------------------------
// Kernel 2: split-K WMMA main loop, N=32 per wave
// ---------------------------------------------------------------------------
__global__ __launch_bounds__(THREADS_PER_BLOCK)
void int8_linear_wmma_splitk(const _Float16*    __restrict__ xf16,
                             const signed char* __restrict__ w,
                             float*             __restrict__ part)
{
    const int tid   = threadIdx.x;
    const int wave  = tid >> 5;
    const int lane  = tid & 31;
    const int ncol  = lane & 15;
    const int khalf = lane >> 4;       // fragment K-half per lane group
    const int klo   = khalf * 8;

    const int nbase = (blockIdx.x * WAVES_PER_BLOCK + wave) * N_PER_WAVE;
    const int n0    = nbase + ncol;         // N tile 0 column
    const int n1    = nbase + 16 + ncol;    // N tile 1 column
    const int split = blockIdx.y;
    const int k0    = split * K_CHUNK;

    const _Float16*    xr0 = xf16 + (size_t)ncol        * K_DIM + k0;  // M 0..15
    const _Float16*    xr1 = xf16 + (size_t)(16 + ncol) * K_DIM + k0;  // M 16..31
    const signed char* wr0 = w    + (size_t)n0 * K_DIM + k0;
    const signed char* wr1 = w    + (size_t)n1 * K_DIM + k0;

    v8f c00 = {};   // (M tile 0, N tile 0)
    v8f c10 = {};   // (M tile 1, N tile 0)
    v8f c01 = {};   // (M tile 0, N tile 1)
    v8f c11 = {};   // (M tile 1, N tile 1)

    union AV { v16h v; uint4 u[2]; };
    union BV { v16h v; unsigned int u[8]; };
    const unsigned int EXP    = 0x64646464u;   // f16 exponent bytes: value 1024+u
    const unsigned int SGN    = 0x80808080u;   // int8 -> biased u8
    const unsigned int SEL_LO = 0x05010400u;   // {0x64, u.b1, 0x64, u.b0}
    const unsigned int SEL_HI = 0x07030602u;   // {0x64, u.b3, 0x64, u.b2}

    #pragma unroll 2
    for (int kb = 0; kb < K_CHUNK; kb += 32) {
        // fire-and-forget stream-ahead for both weight row groups
        __builtin_prefetch(wr0 + kb + 2048, 0, 0);   // global_prefetch_b8
        __builtin_prefetch(wr1 + kb + 2048, 0, 0);

        // A fragments (f16 16x32): halves 0..7 = K(klo..+7), 8..15 = K(16+klo..+7)
        AV a0, a1;
        a0.u[0] = *(const uint4*)(xr0 + kb + klo);
        a0.u[1] = *(const uint4*)(xr0 + kb + 16 + klo);
        a1.u[0] = *(const uint4*)(xr1 + kb + klo);
        a1.u[1] = *(const uint4*)(xr1 + kb + 16 + klo);

        // B fragments: 16 int8 per lane per N tile -> 16 exact f16 (= v + 1152)
        uint2 p0 = *(const uint2*)(wr0 + kb + klo);
        uint2 p1 = *(const uint2*)(wr0 + kb + 16 + klo);
        uint2 q0 = *(const uint2*)(wr1 + kb + klo);
        uint2 q1 = *(const uint2*)(wr1 + kb + 16 + klo);
        p0.x ^= SGN; p0.y ^= SGN; p1.x ^= SGN; p1.y ^= SGN;
        q0.x ^= SGN; q0.y ^= SGN; q1.x ^= SGN; q1.y ^= SGN;

        BV b0, b1;
        b0.u[0] = __builtin_amdgcn_perm(EXP, p0.x, SEL_LO);
        b0.u[1] = __builtin_amdgcn_perm(EXP, p0.x, SEL_HI);
        b0.u[2] = __builtin_amdgcn_perm(EXP, p0.y, SEL_LO);
        b0.u[3] = __builtin_amdgcn_perm(EXP, p0.y, SEL_HI);
        b0.u[4] = __builtin_amdgcn_perm(EXP, p1.x, SEL_LO);
        b0.u[5] = __builtin_amdgcn_perm(EXP, p1.x, SEL_HI);
        b0.u[6] = __builtin_amdgcn_perm(EXP, p1.y, SEL_LO);
        b0.u[7] = __builtin_amdgcn_perm(EXP, p1.y, SEL_HI);

        b1.u[0] = __builtin_amdgcn_perm(EXP, q0.x, SEL_LO);
        b1.u[1] = __builtin_amdgcn_perm(EXP, q0.x, SEL_HI);
        b1.u[2] = __builtin_amdgcn_perm(EXP, q0.y, SEL_LO);
        b1.u[3] = __builtin_amdgcn_perm(EXP, q0.y, SEL_HI);
        b1.u[4] = __builtin_amdgcn_perm(EXP, q1.x, SEL_LO);
        b1.u[5] = __builtin_amdgcn_perm(EXP, q1.x, SEL_HI);
        b1.u[6] = __builtin_amdgcn_perm(EXP, q1.y, SEL_LO);
        b1.u[7] = __builtin_amdgcn_perm(EXP, q1.y, SEL_HI);

        c00 = __builtin_amdgcn_wmma_f32_16x16x32_f16(false, a0.v, false, b0.v,
                                                     (short)0, c00, false, false);
        c10 = __builtin_amdgcn_wmma_f32_16x16x32_f16(false, a1.v, false, b0.v,
                                                     (short)0, c10, false, false);
        c01 = __builtin_amdgcn_wmma_f32_16x16x32_f16(false, a0.v, false, b1.v,
                                                     (short)0, c01, false, false);
        c11 = __builtin_amdgcn_wmma_f32_16x16x32_f16(false, a1.v, false, b1.v,
                                                     (short)0, c11, false, false);
    }

    // f32 partials: one exclusive writer per (split, m, n) -> no atomics needed
    float* p = part + (size_t)split * (M_DIM * N_DIM);
    const int mbase = khalf * 8;
    #pragma unroll
    for (int r = 0; r < 8; ++r) {
        p[(size_t)(mbase + r)      * N_DIM + n0] = c00[r];
        p[(size_t)(16 + mbase + r) * N_DIM + n0] = c10[r];
        p[(size_t)(mbase + r)      * N_DIM + n1] = c01[r];
        p[(size_t)(16 + mbase + r) * N_DIM + n1] = c11[r];
    }
}

// ---------------------------------------------------------------------------
// Kernel 3: deterministic fixed-order reduce + dequant-bias fix + scale + bias
// ---------------------------------------------------------------------------
__global__ __launch_bounds__(256)
void reduce_kernel(const float* __restrict__ part,
                   const float* __restrict__ xsum,
                   const float* __restrict__ scale,
                   const __hip_bfloat16* __restrict__ bias,
                   __hip_bfloat16* __restrict__ out)
{
    const int idx = blockIdx.x * 256 + threadIdx.x;   // 0 .. 32*8192-1
    const int m = idx >> 13;
    const int n = idx & (N_DIM - 1);

    float s = 0.0f;
    #pragma unroll
    for (int sp = 0; sp < SPLITK; ++sp)               // fixed order: deterministic
        s += part[(size_t)sp * M_DIM * N_DIM + idx];
    s -= 1152.0f * xsum[m];                 // remove folded dequant bias (+1152/elem)

    out[idx] = (__hip_bfloat16)(s * scale[0] + (float)bias[n]);
}

// ---------------------------------------------------------------------------
// Fallback (small workspace): single-kernel bf16 dequant WMMA
// ---------------------------------------------------------------------------
__global__ __launch_bounds__(THREADS_PER_BLOCK)
void int8_linear_wmma_fallback(const __hip_bfloat16* __restrict__ x,
                               const signed char*    __restrict__ w,
                               const float*          __restrict__ scale,
                               const __hip_bfloat16* __restrict__ bias,
                               __hip_bfloat16*       __restrict__ out)
{
    const int tid   = threadIdx.x;
    const int wave  = tid >> 5;
    const int lane  = tid & 31;
    const int ncol  = lane & 15;
    const int khalf = lane >> 4;
    const int klo   = khalf * 8;

    const int nbase = (blockIdx.x * WAVES_PER_BLOCK + wave) * 16;
    const int n     = nbase + ncol;

    const __hip_bfloat16* xr0 = x + (size_t)ncol        * K_DIM;
    const __hip_bfloat16* xr1 = x + (size_t)(16 + ncol) * K_DIM;
    const signed char*    wr  = w + (size_t)n * K_DIM;

    v8f c0 = {}; v8f c1 = {};
    union AV { v16bf v; uint4 u[2]; };

    #pragma unroll 2
    for (int kb = 0; kb < K_DIM; kb += 32) {
        __builtin_prefetch(wr + kb + 1024, 0, 0);
        AV a0, a1;
        a0.u[0] = *(const uint4*)(xr0 + kb + klo);
        a0.u[1] = *(const uint4*)(xr0 + kb + 16 + klo);
        a1.u[0] = *(const uint4*)(xr1 + kb + klo);
        a1.u[1] = *(const uint4*)(xr1 + kb + 16 + klo);

        const int2 r0 = *(const int2*)(wr + kb + klo);
        const int2 r1 = *(const int2*)(wr + kb + 16 + klo);

        v16bf b;
        #pragma unroll
        for (int j = 0; j < 4; ++j) {
            b[j]      = (__bf16)(float)((signed char)(r0.x >> (8 * j)));
            b[4 + j]  = (__bf16)(float)((signed char)(r0.y >> (8 * j)));
            b[8 + j]  = (__bf16)(float)((signed char)(r1.x >> (8 * j)));
            b[12 + j] = (__bf16)(float)((signed char)(r1.y >> (8 * j)));
        }
        c0 = __builtin_amdgcn_wmma_f32_16x16x32_bf16(false, a0.v, false, b,
                                                     (short)0, c0, false, false);
        c1 = __builtin_amdgcn_wmma_f32_16x16x32_bf16(false, a1.v, false, b,
                                                     (short)0, c1, false, false);
    }

    const float sc = scale[0];
    const float bv = (float)bias[n];
    const int mbase = khalf * 8;
    #pragma unroll
    for (int r = 0; r < 8; ++r) {
        const int m = mbase + r;
        out[(size_t)m        * N_DIM + n] = (__hip_bfloat16)(c0[r] * sc + bv);
        out[(size_t)(16 + m) * N_DIM + n] = (__hip_bfloat16)(c1[r] * sc + bv);
    }
}

// ---------------------------------------------------------------------------
extern "C" void kernel_launch(void* const* d_in, const int* in_sizes, int n_in,
                              void* d_out, int out_size, void* d_ws, size_t ws_size,
                              hipStream_t stream) {
    (void)in_sizes; (void)n_in; (void)out_size;

    const __hip_bfloat16* x     = (const __hip_bfloat16*)d_in[0];
    const signed char*    w     = (const signed char*)   d_in[1];
    const float*          scale = (const float*)         d_in[2];
    const __hip_bfloat16* bias  = (const __hip_bfloat16*)d_in[3];
    __hip_bfloat16*       out   = (__hip_bfloat16*)      d_out;

    if (d_ws != nullptr && ws_size >= WS_NEEDED) {
        char* ws = (char*)d_ws;
        _Float16* xf16 = (_Float16*)(ws + WS_XF16);
        float*    xsum = (float*)   (ws + WS_XSUM);
        float*    part = (float*)   (ws + WS_PART);

        prep_kernel<<<M_DIM, 256, 0, stream>>>(x, xf16, xsum);

        // (64, 8) blocks x 4 waves = 2048 waves, each: N=32 cols, K-chunk=1024
        dim3 grid(N_DIM / (WAVES_PER_BLOCK * N_PER_WAVE), SPLITK);
        int8_linear_wmma_splitk<<<grid, THREADS_PER_BLOCK, 0, stream>>>(xf16, w, part);

        reduce_kernel<<<(M_DIM * N_DIM) / 256, 256, 0, stream>>>(part, xsum, scale,
                                                                 bias, out);
    } else {
        dim3 grid(N_DIM / (WAVES_PER_BLOCK * 16));
        int8_linear_wmma_fallback<<<grid, THREADS_PER_BLOCK, 0, stream>>>(x, w, scale,
                                                                          bias, out);
    }
}
